// DynamicGlobalWindowTransformer_32066225832467
// MI455X (gfx1250) — compile-verified, matrix operand order checked by reading, not run on hardware
//
#include <hip/hip_runtime.h>
#include <hip/hip_bf16.h>

typedef __bf16 bf16_t;
typedef __attribute__((ext_vector_type(16))) __bf16 v16bf;
typedef __attribute__((ext_vector_type(4)))  __bf16 v4bf;
typedef __attribute__((ext_vector_type(8)))  float  v8f;

#define DIMD   256
#define HEADS  4
#define HD     64
#define FFND   1024
#define BATCH  8
#define SEQ    2048
#define NTOK   (BATCH*SEQ)
#define NEGV   (-1e9f)

// ---------------------------------------------------------------- fragments
// A/B fragment (16-bit, 16x32): lane r = row (M for A, N for B), element e ->
// K = (e%8) + 8*(lane/16) + 16*(e/8).  Base must point at (row, k0).
__device__ __forceinline__ v16bf load_frag_bf16(const bf16_t* base) {
  const int half = (threadIdx.x >> 4) & 1;
  const bf16_t* p = base + 8 * half;
  v16bf f;
#pragma unroll
  for (int e = 0; e < 8; ++e) {
    f[e]     = p[e];
    f[e + 8] = p[e + 16];
  }
  return f;
}

__device__ __forceinline__ v8f wmma_bf16(v16bf a, v16bf b, v8f c) {
  return __builtin_amdgcn_wmma_f32_16x16x32_bf16(
      /*neg_a=*/false, a, /*neg_b=*/false, b,
      /*c_mod=*/(short)0, c, /*reuse_a=*/false, /*reuse_b=*/false);
}

// ---------------------------------------------------------------- cvt
// Pre-convert f32 tensors to bf16 so GEMM fragments are raw b128 loads.
__global__ __launch_bounds__(256) void cvt_bf16(const float* __restrict__ in,
                                                bf16_t* __restrict__ out) {
  const size_t i = (size_t)blockIdx.x * 256 + threadIdx.x;
  const float4 v = ((const float4*)in)[i];
  v4bf o;
  o[0] = (bf16_t)v.x; o[1] = (bf16_t)v.y; o[2] = (bf16_t)v.z; o[3] = (bf16_t)v.w;
  ((v4bf*)out)[i] = o;
}

// ---------------------------------------------------------------- GEMM
// Y[M,N] = A[M,K] @ B[N,K]^T + bias, bf16 inputs, f32 accumulate.
// MODE 0: store f32 to Yf.          MODE 1: ReLU, store bf16 to Yb.
// MODE 2: QKV scatter to bf16 Q(B,H,T,64), K(B,H,T,64), Vt(B,H,64,T).
// Wave computes 32x64 (8 WMMAs / 6 fragment loads per k-step);
// block = 4 waves -> 128x64 tile.
template <int MODE>
__global__ __launch_bounds__(128) void gemm_wmma(
    const bf16_t* __restrict__ A, const bf16_t* __restrict__ B,
    const float* __restrict__ bias, float* __restrict__ Yf,
    bf16_t* __restrict__ Yb,
    bf16_t* __restrict__ Qb, bf16_t* __restrict__ Kb, bf16_t* __restrict__ Vt,
    int M, int N, int K) {
  const int lane = threadIdx.x & 31;
  const int wave = threadIdx.x >> 5;
  const int r    = lane & 15;
  const int half = lane >> 4;
  const int row0 = blockIdx.y * 128 + wave * 32;
  const int col0 = blockIdx.x * 64;

  v8f acc[2][4] = {};
  const bf16_t* ap0 = A + (size_t)(row0 + r) * K;
  const bf16_t* ap1 = A + (size_t)(row0 + 16 + r) * K;
  const bf16_t* bp  = B + (size_t)(col0 + r) * K;

  for (int k0 = 0; k0 < K; k0 += 32) {
    __builtin_prefetch(ap0 + k0 + 64, 0, 0);
    __builtin_prefetch(ap1 + k0 + 64, 0, 0);
    v16bf a0 = load_frag_bf16(ap0 + k0);
    v16bf a1 = load_frag_bf16(ap1 + k0);
#pragma unroll
    for (int nb = 0; nb < 4; ++nb) {
      v16bf bfr = load_frag_bf16(bp + (size_t)nb * 16 * K + k0);
      acc[0][nb] = wmma_bf16(a0, bfr, acc[0][nb]);
      acc[1][nb] = wmma_bf16(a1, bfr, acc[1][nb]);
    }
  }

#pragma unroll
  for (int mi = 0; mi < 2; ++mi) {
#pragma unroll
    for (int nb = 0; nb < 4; ++nb) {
#pragma unroll
      for (int i = 0; i < 8; ++i) {
        const int row = row0 + mi * 16 + i + 8 * half;
        const int col = col0 + nb * 16 + r;
        float y = acc[mi][nb][i] + bias[col];
        if (MODE == 0) {
          Yf[(size_t)row * N + col] = y;
        } else if (MODE == 1) {
          Yb[(size_t)row * N + col] = (bf16_t)fmaxf(y, 0.0f);
        } else {
          const int b = row >> 11, t = row & (SEQ - 1);
          if (col < DIMD) {
            const int h = col >> 6, d = col & 63;
            Qb[(((size_t)b * HEADS + h) * SEQ + t) * HD + d] = (bf16_t)y;
          } else if (col < 2 * DIMD) {
            const int c = col - DIMD, h = c >> 6, d = c & 63;
            Kb[(((size_t)b * HEADS + h) * SEQ + t) * HD + d] = (bf16_t)y;
          } else {
            const int c = col - 2 * DIMD, h = c >> 6, d = c & 63;
            Vt[(((size_t)b * HEADS + h) * HD + d) * SEQ + t] = (bf16_t)y;
          }
        }
      }
    }
  }
}

// ---------------------------------------------------------------- attention
// One wave per (b, h, 32-query tile): two 16-row sub-tiles share every K/V
// fragment.  Two-pass softmax; WMMA for QK^T and PV.  Accumulates per-key
// column sums of P into halt[b,key]; writes O as bf16.
__global__ __launch_bounds__(32) void attn_wmma(
    const bf16_t* __restrict__ Qb, const bf16_t* __restrict__ Kb,
    const bf16_t* __restrict__ Vt, const float* __restrict__ mask,
    bf16_t* __restrict__ attn, float* __restrict__ halt) {
  __shared__ bf16_t lds_p[2][16 * 32];

  const int bid  = blockIdx.x;
  const int qt   = bid & (SEQ / 32 - 1);  // 64 q-tiles of 32 queries
  const int bh   = bid >> 6;
  const int b    = bh >> 2;
  const int lane = threadIdx.x & 31;
  const int r    = lane & 15;
  const int half = lane >> 4;
  const float scale = 0.125f;  // 1/sqrt(64)

  const bf16_t* Qbase = Qb + ((size_t)bh * SEQ + qt * 32) * HD;
  const bf16_t* Kbase = Kb + (size_t)bh * SEQ * HD;
  const bf16_t* Vbase = Vt + (size_t)bh * HD * SEQ;
  const float*  mrow  = mask + (size_t)b * SEQ;

  // Q fragments: 2 sub-tiles x (hd split into two k=32 chunks)
  v16bf qa[2][2];
#pragma unroll
  for (int qi = 0; qi < 2; ++qi) {
    qa[qi][0] = load_frag_bf16(Qbase + (size_t)(qi * 16 + r) * HD + 0);
    qa[qi][1] = load_frag_bf16(Qbase + (size_t)(qi * 16 + r) * HD + 32);
  }

  // ---- pass 1: per-lane online (m,l) over this lane's column subset
  float m[2][8], l[2][8];
#pragma unroll
  for (int qi = 0; qi < 2; ++qi)
#pragma unroll
    for (int i = 0; i < 8; ++i) { m[qi][i] = -3.0e38f; l[qi][i] = 0.0f; }

  for (int kt = 0; kt < SEQ / 16; ++kt) {
    v16bf kb0 = load_frag_bf16(Kbase + (size_t)(kt * 16 + r) * HD + 0);
    v16bf kb1 = load_frag_bf16(Kbase + (size_t)(kt * 16 + r) * HD + 32);
    const float mk = mrow[kt * 16 + r];
#pragma unroll
    for (int qi = 0; qi < 2; ++qi) {
      v8f s = {};
      s = wmma_bf16(qa[qi][0], kb0, s);
      s = wmma_bf16(qa[qi][1], kb1, s);
#pragma unroll
      for (int i = 0; i < 8; ++i) {
        const float sv = (mk > 0.5f) ? NEGV : s[i] * scale;
        const float mn = fmaxf(m[qi][i], sv);
        l[qi][i] = l[qi][i] * __expf(m[qi][i] - mn) + __expf(sv - mn);
        m[qi][i] = mn;
      }
    }
  }
  // merge (m,l) across the 16 lanes of each half (xor 1,2,4,8 stays in-half)
#pragma unroll
  for (int d = 1; d <= 8; d <<= 1) {
#pragma unroll
    for (int qi = 0; qi < 2; ++qi) {
#pragma unroll
      for (int i = 0; i < 8; ++i) {
        const float mo = __shfl_xor(m[qi][i], d, 32);
        const float lo = __shfl_xor(l[qi][i], d, 32);
        const float mn = fmaxf(m[qi][i], mo);
        l[qi][i] = l[qi][i] * __expf(m[qi][i] - mn) + lo * __expf(mo - mn);
        m[qi][i] = mn;
      }
    }
  }
  float rl[2][8];
#pragma unroll
  for (int qi = 0; qi < 2; ++qi)
#pragma unroll
    for (int i = 0; i < 8; ++i) rl[qi][i] = 1.0f / l[qi][i];

  // ---- pass 2: P = exp(s-m)/l ; O += P @ V ; colsums -> halt
  v8f o[2][4] = {};
  for (int kt2 = 0; kt2 < SEQ / 32; ++kt2) {
#pragma unroll
    for (int sub = 0; sub < 2; ++sub) {
      const int kt = kt2 * 2 + sub;
      v16bf kb0 = load_frag_bf16(Kbase + (size_t)(kt * 16 + r) * HD + 0);
      v16bf kb1 = load_frag_bf16(Kbase + (size_t)(kt * 16 + r) * HD + 32);
      const float mk = mrow[kt * 16 + r];
      float cs = 0.0f;
#pragma unroll
      for (int qi = 0; qi < 2; ++qi) {
        v8f s = {};
        s = wmma_bf16(qa[qi][0], kb0, s);
        s = wmma_bf16(qa[qi][1], kb1, s);
#pragma unroll
        for (int i = 0; i < 8; ++i) {
          const float sv = (mk > 0.5f) ? NEGV : s[i] * scale;
          const float p  = __expf(sv - m[qi][i]) * rl[qi][i];
          cs += p;
          lds_p[qi][(i + 8 * half) * 32 + sub * 16 + r] = (bf16_t)p;
        }
      }
      cs += __shfl_xor(cs, 16, 32);
      if (lane < 16) atomicAdd(&halt[(size_t)b * SEQ + kt * 16 + lane], cs);
    }
    __syncthreads();
    v16bf pa0 = load_frag_bf16(lds_p[0] + r * 32);  // A frag: 16 q x 32 keys
    v16bf pa1 = load_frag_bf16(lds_p[1] + r * 32);
#pragma unroll
    for (int nb = 0; nb < 4; ++nb) {
      v16bf vb = load_frag_bf16(Vbase + (size_t)(nb * 16 + r) * SEQ + kt2 * 32);
      o[0][nb] = wmma_bf16(pa0, vb, o[0][nb]);
      o[1][nb] = wmma_bf16(pa1, vb, o[1][nb]);
    }
    __syncthreads();
  }

  // store O into attn (B,T,D) bf16 (only consumed by the Wo GEMM)
  const int h = bh & 3;
#pragma unroll
  for (int qi = 0; qi < 2; ++qi) {
#pragma unroll
    for (int nb = 0; nb < 4; ++nb) {
#pragma unroll
      for (int i = 0; i < 8; ++i) {
        const int t = qt * 32 + qi * 16 + i + 8 * half;
        attn[((size_t)b * SEQ + t) * DIMD + h * HD + nb * 16 + r] =
            (bf16_t)o[qi][nb][i];
      }
    }
  }
}

// ---------------------------------------------------------------- LN
// out = LN(A + Bv); optionally also emit a bf16 copy for the next GEMM.
template <bool WRITE_BF16>
__global__ __launch_bounds__(256) void add_ln(
    const float* __restrict__ A, const float* __restrict__ Bv,
    const float* __restrict__ g, const float* __restrict__ be,
    float* __restrict__ out, bf16_t* __restrict__ outb) {
  __shared__ float red[256];
  const int row = blockIdx.x, tid = threadIdx.x;
  const float v = A[(size_t)row * DIMD + tid] + Bv[(size_t)row * DIMD + tid];
  red[tid] = v;
  __syncthreads();
  for (int s = 128; s > 0; s >>= 1) {
    if (tid < s) red[tid] += red[tid + s];
    __syncthreads();
  }
  const float mu = red[0] * (1.0f / DIMD);
  __syncthreads();
  const float d = v - mu;
  red[tid] = d * d;
  __syncthreads();
  for (int s = 128; s > 0; s >>= 1) {
    if (tid < s) red[tid] += red[tid + s];
    __syncthreads();
  }
  const float var = red[0] * (1.0f / DIMD);
  const float y = d * rsqrtf(var + 1e-5f) * g[tid] + be[tid];
  out[(size_t)row * DIMD + tid] = y;
  if (WRITE_BF16) outb[(size_t)row * DIMD + tid] = (bf16_t)y;
}

// ---------------------------------------------------------------- halting
__global__ __launch_bounds__(256) void halt_finalize(
    const float* __restrict__ hacc, const float* __restrict__ mask,
    float* __restrict__ out) {
  __shared__ float red[256];
  const int b = blockIdx.x, tid = threadIdx.x;
  float lm = -3.0e38f;
  for (int j = tid; j < SEQ; j += 256) {
    const float v = hacc[(size_t)b * SEQ + j] * 0.25f +
                    mask[(size_t)b * SEQ + j] * (-1e10f);
    lm = fmaxf(lm, v);
  }
  red[tid] = lm;
  __syncthreads();
  for (int s = 128; s > 0; s >>= 1) {
    if (tid < s) red[tid] = fmaxf(red[tid], red[tid + s]);
    __syncthreads();
  }
  const float mx = red[0];
  __syncthreads();
  float ls = 0.0f;
  for (int j = tid; j < SEQ; j += 256) {
    const float v = hacc[(size_t)b * SEQ + j] * 0.25f +
                    mask[(size_t)b * SEQ + j] * (-1e10f);
    ls += __expf(v - mx);
  }
  red[tid] = ls;
  __syncthreads();
  for (int s = 128; s > 0; s >>= 1) {
    if (tid < s) red[tid] += red[tid + s];
    __syncthreads();
  }
  const float inv = 1.0f / red[0];
  for (int j = tid; j < SEQ; j += 256) {
    const float v = hacc[(size_t)b * SEQ + j] * 0.25f +
                    mask[(size_t)b * SEQ + j] * (-1e10f);
    out[(size_t)b * SEQ + j] = __expf(v - mx) * inv;
  }
}

// ---------------------------------------------------------------- launch
extern "C" void kernel_launch(void* const* d_in, const int* in_sizes, int n_in,
                              void* d_out, int out_size, void* d_ws,
                              size_t ws_size, hipStream_t stream) {
  (void)in_sizes; (void)n_in; (void)out_size; (void)ws_size;
  const float* x    = (const float*)d_in[0];
  const float* mask = (const float*)d_in[1];
  const float* Wqkv = (const float*)d_in[2];
  const float* bqkv = (const float*)d_in[3];
  const float* Wo   = (const float*)d_in[4];
  const float* bo   = (const float*)d_in[5];
  const float* W1   = (const float*)d_in[6];
  const float* b1   = (const float*)d_in[7];
  const float* W2   = (const float*)d_in[8];
  const float* b2   = (const float*)d_in[9];
  const float* g1   = (const float*)d_in[10];
  const float* be1  = (const float*)d_in[11];
  const float* g2   = (const float*)d_in[12];
  const float* be2  = (const float*)d_in[13];

  char* ws = (char*)d_ws;
  size_t off = 0;
  auto alloc = [&](size_t bytes) {
    char* p = ws + off;
    off += (bytes + 255) & ~(size_t)255;
    return p;
  };
  bf16_t* xb    = (bf16_t*)alloc((size_t)NTOK * DIMD * 2);
  bf16_t* Wqkvb = (bf16_t*)alloc((size_t)3 * DIMD * DIMD * 2);
  bf16_t* Wob   = (bf16_t*)alloc((size_t)DIMD * DIMD * 2);
  bf16_t* W1b   = (bf16_t*)alloc((size_t)FFND * DIMD * 2);
  bf16_t* W2b   = (bf16_t*)alloc((size_t)DIMD * FFND * 2);
  bf16_t* Qb    = (bf16_t*)alloc((size_t)NTOK * DIMD * 2);
  bf16_t* Kb    = (bf16_t*)alloc((size_t)NTOK * DIMD * 2);
  bf16_t* Vt    = (bf16_t*)alloc((size_t)NTOK * DIMD * 2);
  bf16_t* attnb = (bf16_t*)alloc((size_t)NTOK * DIMD * 2);
  float*  attnO = (float*) alloc((size_t)NTOK * DIMD * 4);
  float*  hbuf  = (float*) alloc((size_t)NTOK * DIMD * 4);
  bf16_t* hb    = (bf16_t*)alloc((size_t)NTOK * DIMD * 2);
  bf16_t* ffn1b = (bf16_t*)alloc((size_t)NTOK * FFND * 2);
  float*  ffn2  = (float*) alloc((size_t)NTOK * DIMD * 4);
  float*  halt  = (float*) alloc((size_t)NTOK * 4);

  float* out      = (float*)d_out;
  float* out_halt = out + (size_t)NTOK * DIMD;

  hipMemsetAsync(halt, 0, (size_t)NTOK * sizeof(float), stream);

  // pre-convert activations + weights to bf16 (4 elems / thread)
  cvt_bf16<<<NTOK * DIMD / 1024, 256, 0, stream>>>(x, xb);
  cvt_bf16<<<3 * DIMD * DIMD / 1024, 256, 0, stream>>>(Wqkv, Wqkvb);
  cvt_bf16<<<DIMD * DIMD / 1024, 256, 0, stream>>>(Wo, Wob);
  cvt_bf16<<<FFND * DIMD / 1024, 256, 0, stream>>>(W1, W1b);
  cvt_bf16<<<DIMD * FFND / 1024, 256, 0, stream>>>(W2, W2b);

  // QKV projection -> scatter bf16 Q/K/Vt
  gemm_wmma<2><<<dim3(3 * DIMD / 64, NTOK / 128), 128, 0, stream>>>(
      xb, Wqkvb, bqkv, nullptr, nullptr, Qb, Kb, Vt, NTOK, 3 * DIMD, DIMD);

  // fused attention + halting column sums (bf16 output)
  attn_wmma<<<BATCH * HEADS * (SEQ / 32), 32, 0, stream>>>(
      Qb, Kb, Vt, mask, attnb, halt);

  // output projection (f32 out, feeds residual+LN)
  gemm_wmma<0><<<dim3(DIMD / 64, NTOK / 128), 128, 0, stream>>>(
      attnb, Wob, bo, attnO, nullptr, nullptr, nullptr, nullptr,
      NTOK, DIMD, DIMD);

  // h = LN(x + attn_out); also bf16 copy for FFN1
  add_ln<true><<<NTOK, 256, 0, stream>>>(x, attnO, g1, be1, hbuf, hb);

  // FFN1 with ReLU (bf16 out, feeds FFN2)
  gemm_wmma<1><<<dim3(FFND / 64, NTOK / 128), 128, 0, stream>>>(
      hb, W1b, b1, nullptr, ffn1b, nullptr, nullptr, nullptr,
      NTOK, FFND, DIMD);

  // FFN2 (f32 out, feeds final residual+LN)
  gemm_wmma<0><<<dim3(DIMD / 64, NTOK / 128), 128, 0, stream>>>(
      ffn1b, W2b, b2, ffn2, nullptr, nullptr, nullptr, nullptr,
      NTOK, DIMD, FFND);

  // out = LN(h + ffn)
  add_ln<false><<<NTOK, 256, 0, stream>>>(hbuf, ffn2, g2, be2, out, nullptr);

  // halting softmax
  halt_finalize<<<BATCH, 256, 0, stream>>>(halt, mask, out_halt);
}
